// Model_31301721653322
// MI455X (gfx1250) — matrix operand order repaired and checked by reading, not compiled, not run on hardware
//
#include <hip/hip_runtime.h>
#include <hip/hip_bf16.h>

#define BB 32
#define TT 384
#define DIMD 896
#define HH 128
#define HDD 7
#define WW 5
#define INDIM 1071
#define OUTD 60
#define BIGN 1792
#define BASEC 336

typedef __attribute__((ext_vector_type(16))) __bf16 v16bf;
typedef __attribute__((ext_vector_type(8)))  float  v8f;

// ---------------------------------------------------------------------------
// helpers
// ---------------------------------------------------------------------------
__device__ __forceinline__ unsigned short f2bf(float x) {
  unsigned int u = __float_as_uint(x);
  unsigned int r = u + 0x7FFFu + ((u >> 16) & 1u);   // round-to-nearest-even
  return (unsigned short)(r >> 16);
}

// blockDim.x must be 256
__device__ __forceinline__ float blockSum256(float v, float* red) {
  int tid = threadIdx.x;
  red[tid] = v;
  __syncthreads();
  for (int s = 128; s > 0; s >>= 1) {
    if (tid < s) red[tid] += red[tid + s];
    __syncthreads();
  }
  float r = red[0];
  __syncthreads();
  return r;
}

__device__ __forceinline__ float elu1(float y) {
  return y > 0.f ? y : (expf(y) - 1.f);
}

// ---------------------------------------------------------------------------
// WMMA GEMM:  C[M,N] = A[M,K] (f32, row-major) @ W[K,N] (f32, row-major) + bias
// bf16 staged in LDS (pair-packed, branch-free clamped guards), f32 accum via
// v_wmma_f32_16x16x32_bf16.  128x64 C tile / block, 256 threads = 8 waves;
// each wave owns a 16x64 strip -> 4 WMMA per K-step sharing one A fragment.
// Next tile's global loads overlap the WMMAs (software pipelining).
// ---------------------------------------------------------------------------
#define LDS_LD 34   // 32 + 2 pad (ushorts)

__global__ __launch_bounds__(256)
void gemm_bf16_wmma(const float* __restrict__ A, const float* __restrict__ W,
                    const float* __restrict__ bias, float* __restrict__ C,
                    int M, int N, int K)
{
  __shared__ unsigned short As[128 * LDS_LD];  // 128 m-rows x 32 k
  __shared__ unsigned short Bs[64 * LDS_LD];   // transposed: [n][k]

  const int tid  = threadIdx.x;
  const int wid  = tid >> 5;
  const int lane = tid & 31;
  const int h    = lane >> 4;      // half of wave
  const int l15  = lane & 15;
  const int rowBase = blockIdx.y * 128;
  const int colBase = blockIdx.x * 64;

  const int kTiles = (K + 31) / 32;

  // staging: A tile 128x32 = 2048 pairs (8/thread); W tile 32x64 = 1024 pairs
  // (4/thread).  pair e -> (row = e>>4, kpair = (e&15)*2)
  // branch-free guards: clamp address into bounds, zero value via select.
  auto stageLoad = [&](int kt, unsigned int* ra, unsigned int* rw) {
    const int k0 = kt * 32;
#pragma unroll
    for (int i = 0; i < 8; ++i) {              // ---- A pairs
      int e  = tid + i * 256;
      int gr = rowBase + (e >> 4);
      int gc = k0 + ((e & 15) << 1);
      int grc = gr < M ? gr : M - 1;
      int gc0 = gc < K ? gc : K - 1;
      int gc1 = gc + 1 < K ? gc + 1 : K - 1;
      float a0 = A[(size_t)grc * K + gc0];
      float a1 = A[(size_t)grc * K + gc1];
      bool rok = gr < M;
      a0 = (rok && gc < K)     ? a0 : 0.f;
      a1 = (rok && gc + 1 < K) ? a1 : 0.f;
      ra[i] = (unsigned int)f2bf(a0) | ((unsigned int)f2bf(a1) << 16);
    }
#pragma unroll
    for (int i = 0; i < 4; ++i) {              // ---- W pairs (2 K-rows, 1 col)
      int e  = tid + i * 256;
      int wn = colBase + (e >> 4);
      int wk = k0 + ((e & 15) << 1);
      int wnc = wn < N ? wn : N - 1;
      int wk0 = wk < K ? wk : K - 1;
      int wk1 = wk + 1 < K ? wk + 1 : K - 1;
      float b0 = W[(size_t)wk0 * N + wnc];
      float b1 = W[(size_t)wk1 * N + wnc];
      bool cok = wn < N;
      b0 = (cok && wk < K)     ? b0 : 0.f;
      b1 = (cok && wk + 1 < K) ? b1 : 0.f;
      rw[i] = (unsigned int)f2bf(b0) | ((unsigned int)f2bf(b1) << 16);
    }
  };
  auto stageStore = [&](const unsigned int* ra, const unsigned int* rw) {
#pragma unroll
    for (int i = 0; i < 8; ++i) {
      int e = tid + i * 256;
      *(unsigned int*)&As[(e >> 4) * LDS_LD + ((e & 15) << 1)] = ra[i];
    }
#pragma unroll
    for (int i = 0; i < 4; ++i) {
      int e = tid + i * 256;
      *(unsigned int*)&Bs[(e >> 4) * LDS_LD + ((e & 15) << 1)] = rw[i];
    }
  };

  v8f acc[4] = {v8f{}, v8f{}, v8f{}, v8f{}};

  unsigned int ra[8], rw[4];
  stageLoad(0, ra, rw);

  for (int kt = 0; kt < kTiles; ++kt) {
    stageStore(ra, rw);
    __syncthreads();

    // issue next tile's global loads; they overlap LDS reads + WMMA below
    unsigned int na[8], nw[4];
    if (kt + 1 < kTiles) stageLoad(kt + 1, na, nw);

    // A fragment: 16x32 bf16, per ISA VGPR layout (one per wave)
    union { v16bf v; unsigned int u[8]; } af;
    const int mrow = wid * 16 + l15;
#pragma unroll
    for (int i = 0; i < 8; ++i) {
      int koff = ((i >> 2) << 4) + (h << 3) + ((i & 3) << 1);
      af.u[i] = *(const unsigned int*)&As[mrow * LDS_LD + koff];
    }
    // 4 B fragments: lane = column, VGPR i holds K = (i>>2)*16 + h*8 + 2(i&3), +1
#pragma unroll
    for (int j = 0; j < 4; ++j) {
      union { v16bf v; unsigned int u[8]; } bf;
      const int ncol = j * 16 + l15;
#pragma unroll
      for (int i = 0; i < 8; ++i) {
        int k = ((i >> 2) << 4) + (h << 3) + ((i & 3) << 1);
        bf.u[i] = *(const unsigned int*)&Bs[ncol * LDS_LD + k];
      }
      acc[j] = __builtin_amdgcn_wmma_f32_16x16x32_bf16(false, af.v, false, bf.v,
                                                       (short)0, acc[j], false, false);
    }
    __syncthreads();

#pragma unroll
    for (int i = 0; i < 8; ++i) ra[i] = na[i];
#pragma unroll
    for (int i = 0; i < 4; ++i) rw[i] = nw[i];
  }

  // C/D layout: VGPR r -> M = wid*16 + h*8 + r ; N = j*16 + (lane&15)
#pragma unroll
  for (int j = 0; j < 4; ++j) {
#pragma unroll
    for (int r = 0; r < 8; ++r) {
      int gr = rowBase + wid * 16 + h * 8 + r;
      int gc = colBase + j * 16 + l15;
      if (gr < M && gc < N)
        C[(size_t)gr * N + gc] = acc[j][r] + (bias ? bias[gc] : 0.f);
    }
  }
}

// ---------------------------------------------------------------------------
// Preproc kernel 1: per-batch statistics
// stats[b*16 + {0..2}] = mean_c, {3..5} = rms_cen, {6..8} = rms_stab, {9} = dmean
// ---------------------------------------------------------------------------
__global__ __launch_bounds__(256)
void preproc_stats_kernel(const float* __restrict__ x, float* __restrict__ mt,
                          float* __restrict__ nmask, int* __restrict__ idxs,
                          float* __restrict__ stats)
{
  __shared__ float red[256];
  int b = blockIdx.x;
  const float* xb = x + (size_t)b * TT * 63;

  float sum_x[3] = {0,0,0}, cnt_x[3] = {0,0,0};
  float dsum = 0.f, dcnt = 0.f, vcnt = 0.f;

  for (int t = threadIdx.x; t < TT; t += 256) {
    const float* xt = xb + t * 63;
    float ps[3] = {0,0,0}, pc[3] = {0,0,0};
    bool nz = false;
    for (int p = 0; p < 21; ++p)
      for (int a = 0; a < 3; ++a) {
        float v = xt[p * 3 + a];
        ps[a] += v;
        if (v != 0.f) { pc[a] += 1.f; nz = true; }
      }
    for (int a = 0; a < 3; ++a) {
      sum_x[a] += ps[a]; cnt_x[a] += pc[a];
      mt[((size_t)b * TT + t) * 3 + a] = ps[a] / (pc[a] + 1e-5f);
    }
    nmask[b * TT + t] = nz ? 1.f : 0.f;
    vcnt += nz ? 1.f : 0.f;
    for (int r = 0; r < 20; ++r)
      for (int c = 0; c + r < 20; ++c) {
        float dx = xt[r * 3 + 0] - xt[c * 3 + 0];
        float dy = xt[r * 3 + 1] - xt[c * 3 + 1];
        float ssq = dx * dx + dy * dy;
        float d = (ssq == 0.f) ? 0.f : sqrtf(ssq);
        dsum += d;
        if (d != 0.f) dcnt += 1.f;
      }
  }

  float mean_c[3];
  for (int a = 0; a < 3; ++a) {
    float S = blockSum256(sum_x[a], red);
    float Cn = blockSum256(cnt_x[a], red);
    mean_c[a] = S / (Cn + 1e-5f);
  }
  dsum = blockSum256(dsum, red);
  dcnt = blockSum256(dcnt, red);
  vcnt = blockSum256(vcnt, red);

  float sqc[3] = {0,0,0}, cc[3] = {0,0,0}, sqs[3] = {0,0,0}, cs[3] = {0,0,0};
  for (int t = threadIdx.x; t < TT; t += 256) {
    const float* xt = xb + t * 63;
    const float* m = &mt[((size_t)b * TT + t) * 3];
    for (int p = 0; p < 21; ++p)
      for (int a = 0; a < 3; ++a) {
        float v = xt[p * 3 + a];
        float y = v - mean_c[a];
        sqc[a] += y * y; if (y != 0.f) cc[a] += 1.f;
        float y2 = v - m[a];
        sqs[a] += y2 * y2; if (y2 != 0.f) cs[a] += 1.f;
      }
  }
  float st[10];
  for (int a = 0; a < 3; ++a) {
    float Sq = blockSum256(sqc[a], red), Cq = blockSum256(cc[a], red);
    st[3 + a] = sqrtf(Sq / (Cq + 1e-5f));
    float Ss = blockSum256(sqs[a], red), Cs2 = blockSum256(cs[a], red);
    st[6 + a] = sqrtf(Ss / (Cs2 + 1e-5f));
    st[a] = mean_c[a];
  }
  st[9] = dsum / (dcnt + 1e-5f);
  if (threadIdx.x == 0) {
    for (int i = 0; i < 10; ++i) stats[b * 16 + i] = st[i];
    idxs[b] = (int)(vcnt + 0.5f) - 1;
  }
}

// ---------------------------------------------------------------------------
// Preproc kernel 2: per-(b,t) features -> f[:,:,0:399], base[:,:,0:336]
// ---------------------------------------------------------------------------
__global__ __launch_bounds__(256)
void preproc_feat_kernel(const float* __restrict__ x, const float* __restrict__ mt,
                         const float* __restrict__ stats, float* __restrict__ f,
                         float* __restrict__ base)
{
  int bt = blockIdx.x;
  int b = bt / TT;
  __shared__ float sx[63], sm[3], srms3[3], sst[10];
  int tid = threadIdx.x;
  if (tid < 63) sx[tid] = x[(size_t)bt * 63 + tid];
  if (tid < 3)  sm[tid] = mt[(size_t)bt * 3 + tid];
  if (tid < 10) sst[tid] = stats[b * 16 + tid];
  __syncthreads();
  if (tid < 3) {
    float s = 0.f, c = 0.f;
    for (int p = 0; p < 21; ++p) {
      float y = sx[p * 3 + tid] - sm[tid];
      s += y * y; if (y != 0.f) c += 1.f;
    }
    srms3[tid] = sqrtf(s / (c + 1e-5f));
  }
  __syncthreads();

  float invd = 1.f / (sst[9] + 1e-5f);
  float* fr = f + (size_t)bt * INDIM;
  float* br = base + (size_t)bt * BASEC;

  for (int q = tid; q < 210; q += 256) {
    int r = 0, rem = q;
    while (rem >= 20 - r) { rem -= 20 - r; ++r; }
    int c = rem;
    float dx = sx[r * 3 + 0] - sx[c * 3 + 0];
    float dy = sx[r * 3 + 1] - sx[c * 3 + 1];
    float ssq = dx * dx + dy * dy;
    float d = (ssq == 0.f) ? 0.f : sqrtf(ssq);
    d *= invd;
    fr[q] = d; br[q] = d;
  }
  for (int i = tid; i < 63; i += 256) {
    int a = i % 3;
    float v = sx[i];
    float cen = (v - sst[a])     / (sst[3 + a] + 1e-5f);
    float stb = (v - sm[a])      / (sst[6 + a] + 1e-5f);
    float shp = (v - sm[a])      / (srms3[a]   + 1e-5f);
    fr[210 + i] = cen; fr[273 + i] = stb; fr[336 + i] = shp;
    br[210 + i] = cen; br[273 + i] = stb;
  }
}

// ---------------------------------------------------------------------------
// Preproc kernel 3: prev/next temporal diffs -> f[:,:,399:1071]
// ---------------------------------------------------------------------------
__global__ __launch_bounds__(256)
void preproc_diff_kernel(const float* __restrict__ base, const int* __restrict__ idxs,
                         float* __restrict__ f)
{
  int bt = blockIdx.x;
  int t = bt % TT;
  __shared__ int zflag;
  if (threadIdx.x == 0) {
    int z = 0;
    for (int j = 0; j < BB; ++j)
      if (idxs[j] == t) z = 1;
    zflag = z;
  }
  __syncthreads();
  const float* cur = base + (size_t)bt * BASEC;
  float* fr = f + (size_t)bt * INDIM;
  for (int i = threadIdx.x; i < BASEC; i += 256) {
    float pv = (t > 0) ? (cur[i] - cur[i - BASEC]) : 0.f;
    float nv = (t < TT - 1 && !zflag) ? (cur[i + BASEC] - cur[i]) : 0.f;
    fr[399 + i] = pv;
    fr[735 + i] = nv;
  }
}

// ---------------------------------------------------------------------------
// Fused LayerNorm + ELU.  mode 0: plain; 1: += add[row]; 2: += sinusoidal PE(t)
// ---------------------------------------------------------------------------
__global__ __launch_bounds__(256)
void ln_elu_kernel(const float* __restrict__ X, const float* __restrict__ g,
                   const float* __restrict__ be, const float* __restrict__ add,
                   float* __restrict__ O, int N, int mode)
{
  __shared__ float red[256];
  int row = blockIdx.x;
  const float* xr = X + (size_t)row * N;
  float s = 0.f, sq = 0.f;
  for (int i = threadIdx.x; i < N; i += 256) {
    float v = xr[i];
    s += v; sq += v * v;
  }
  s  = blockSum256(s, red);
  sq = blockSum256(sq, red);
  float mu = s / (float)N;
  float var = sq / (float)N - mu * mu;
  float inv = rsqrtf(var + 1e-5f);
  int t = row % TT;
  for (int i = threadIdx.x; i < N; i += 256) {
    float y = (xr[i] - mu) * inv * g[i] + be[i];
    y = elu1(y);
    if (mode == 1) {
      y += add[(size_t)row * N + i];
    } else if (mode == 2) {
      int j = i >> 1;
      // div_j = exp(-(2j) * ln(10000)/896);  ln(10000)/896 = 0.0102793754
      float ang = (float)t * expf(-(float)(2 * j) * 0.0102793754f);
      y += (i & 1) ? cosf(ang) : sinf(ang);
    }
    O[(size_t)row * N + i] = y;
  }
}

// ---------------------------------------------------------------------------
// ae = exp(ae_pre) * nmask
// ---------------------------------------------------------------------------
__global__ __launch_bounds__(256)
void ae_exp_kernel(float* __restrict__ ae, const float* __restrict__ nmask, int n)
{
  int i = blockIdx.x * 256 + threadIdx.x;
  if (i < n) ae[i] = expf(ae[i]) * nmask[i / HH];
}

// colsum[b,h] = sum_t ae[b,t,h]
__global__ __launch_bounds__(128)
void colsum_kernel(const float* __restrict__ ae, float* __restrict__ colsum)
{
  int b = blockIdx.x, h = threadIdx.x;
  float s = 0.f;
  for (int t = 0; t < TT; ++t) s += ae[((size_t)b * TT + t) * HH + h];
  colsum[b * HH + h] = s;
}

// gp[b,h*7+d] = sum_t ae[b,t,h]*v[b,t,h*7+d] / (colsum[b,h] + 1e-5)
__global__ __launch_bounds__(128)
void gp_kernel(const float* __restrict__ ae, const float* __restrict__ v,
               const float* __restrict__ colsum, float* __restrict__ gp)
{
  int b = blockIdx.x, h = threadIdx.x;
  float acc[HDD] = {0,0,0,0,0,0,0};
  for (int t = 0; t < TT; ++t) {
    float a = ae[((size_t)b * TT + t) * HH + h];
    const float* vp = v + ((size_t)b * TT + t) * DIMD + h * HDD;
#pragma unroll
    for (int d = 0; d < HDD; ++d) acc[d] += a * vp[d];
  }
  float inv = 1.f / (colsum[b * HH + h] + 1e-5f);
#pragma unroll
  for (int d = 0; d < HDD; ++d) gp[(size_t)b * DIMD + h * HDD + d] = acc[d] * inv;
}

// in-place exp
__global__ __launch_bounds__(256)
void expv_kernel(float* __restrict__ p, int n)
{
  int i = blockIdx.x * 256 + threadIdx.x;
  if (i < n) p[i] = expf(p[i]);
}

// ---------------------------------------------------------------------------
// Window combine (faithful to reference slot pairing):
//   slots s0 = pos0,  s_{j+1} = ae(window j) * pos_{j+1}   (j = 0..4)
//   denom = sum(all 6 slots) + 1e-5
//   out = sum_{j=0..4} (s_j/denom) * v(window j)  +  (s0/denom) * gp
// ---------------------------------------------------------------------------
__global__ __launch_bounds__(128)
void combine_kernel(const float* __restrict__ ae, const float* __restrict__ v,
                    const float* __restrict__ gp, const float* __restrict__ pos,
                    float* __restrict__ out, int dil)
{
  int bt = blockIdx.x;
  int b = bt / TT, t = bt % TT;
  int h = threadIdx.x;

  int   p[WW];
  float ew[WW];
#pragma unroll
  for (int j = 0; j < WW; ++j) {
    int pp = t + dil * (j - WW / 2);
    bool valid = (pp >= 0) && (pp < TT);
    p[j] = valid ? pp : -1;
    ew[j] = valid ? ae[((size_t)b * TT + pp) * HH + h] : 0.f;
  }
  float s[WW + 1];
  s[0] = pos[(size_t)b * (WW + 1) * HH + 0 * HH + h];
  float denom = s[0];
#pragma unroll
  for (int j = 0; j < WW; ++j) {
    s[j + 1] = ew[j] * pos[(size_t)b * (WW + 1) * HH + (j + 1) * HH + h];
    denom += s[j + 1];
  }
  denom += 1e-5f;
  float invd = 1.f / denom;

  float o[HDD];
  const float* gpp = gp + (size_t)b * DIMD + h * HDD;
  float a0 = s[0] * invd;
#pragma unroll
  for (int d = 0; d < HDD; ++d) o[d] = a0 * gpp[d];
#pragma unroll
  for (int j = 0; j < WW; ++j) {
    if (p[j] < 0) continue;                 // zero-padded window row
    float a = s[j] * invd;                  // note: slot j pairs window j
    const float* vp = v + ((size_t)b * TT + p[j]) * DIMD + h * HDD;
#pragma unroll
    for (int d = 0; d < HDD; ++d) o[d] += a * vp[d];
  }
  float* op = out + (size_t)bt * DIMD + h * HDD;
#pragma unroll
  for (int d = 0; d < HDD; ++d) op[d] = o[d];
}

// ---------------------------------------------------------------------------
// host launcher
// ---------------------------------------------------------------------------
extern "C" void kernel_launch(void* const* d_in, const int* in_sizes, int n_in,
                              void* d_out, int out_size, void* d_ws, size_t ws_size,
                              hipStream_t stream) {
  (void)out_size; (void)ws_size; (void)n_in;
  // input index maps: insertion-order vs jax-sorted-key flatten
  int xi, w1, b1, g1, be1, w2, b2, g2, be2;
  int aw, ab, p1w, p1b, plg, plb, p2w, p2b, vw, vb, vlg, vlb, ow, ob, olg, olb;
  int outw, outb;
  if (in_sizes[0] == BB * TT * 21 * 3) {  // insertion order: x first
    xi = 0;
    w1 = 1; b1 = 2; g1 = 3; be1 = 4; w2 = 5; b2 = 6; g2 = 7; be2 = 8;
    aw = 9; ab = 10; p1w = 11; p1b = 12; plg = 13; plb = 14; p2w = 15; p2b = 16;
    vw = 17; vb = 18; vlg = 19; vlb = 20; ow = 21; ob = 22; olg = 23; olb = 24;
    outw = 25; outb = 26;
  } else {                                // jax pytree sorted keys, x last
    b1 = 0; b2 = 1; be1 = 2; be2 = 3; g1 = 4; g2 = 5; w1 = 6; w2 = 7;
    ab = 8; aw = 9; ob = 10; ow = 11; olb = 12; olg = 13; p1b = 14; p1w = 15;
    p2b = 16; p2w = 17; plb = 18; plg = 19; vb = 20; vw = 21; vlb = 22; vlg = 23;
    outb = 24; outw = 25; xi = 26;
  }
  auto F = [&](int i) { return (const float*)d_in[i]; };
  const float* x = F(xi);

  const size_t MT = (size_t)BB * TT;     // 12288 rows
  float* ws = (float*)d_ws;
  size_t off = 0;
  auto alloc = [&](size_t n) { float* p = ws + off; off += n; return p; };
  float* fbuf  = alloc(MT * INDIM);
  float* big   = alloc(MT * BIGN);       // universal pre-LN scratch
  float* hbuf  = alloc(MT * DIMD);       // running activations
  float* vbuf  = alloc(MT * DIMD);
  float* obuf  = alloc(MT * DIMD);
  float* aebuf = alloc(MT * HH);
  float* baseb = alloc(MT * BASEC);
  float* mtb   = alloc(MT * 3);
  float* nmk   = alloc(MT);
  float* statb = alloc((size_t)BB * 16);
  float* colsb = alloc((size_t)BB * HH);
  float* gpb   = alloc((size_t)BB * DIMD);
  float* sm1   = alloc((size_t)BB * HH);
  float* sm2   = alloc((size_t)BB * HH);
  float* posb  = alloc((size_t)BB * (WW + 1) * HH);
  int*   idxb  = (int*)alloc(BB);

  // ---- preprocessing -> fbuf [12288,1071]
  preproc_stats_kernel<<<BB, 256, 0, stream>>>(x, mtb, nmk, idxb, statb);
  preproc_feat_kernel<<<(unsigned)MT, 256, 0, stream>>>(x, mtb, statb, fbuf, baseb);
  preproc_diff_kernel<<<(unsigned)MT, 256, 0, stream>>>(baseb, idxb, fbuf);

  const unsigned MB = (unsigned)((MT + 127) / 128);  // 96 row-blocks
  // ---- input MLP
  {
    dim3 gin1((BIGN + 63) / 64, MB);
    gemm_bf16_wmma<<<gin1, 256, 0, stream>>>(fbuf, F(w1), F(b1), big, (int)MT, BIGN, INDIM);
    ln_elu_kernel<<<(unsigned)MT, 256, 0, stream>>>(big, F(g1), F(be1), nullptr, big, BIGN, 0);
    dim3 gin2((DIMD + 63) / 64, MB);
    gemm_bf16_wmma<<<gin2, 256, 0, stream>>>(big, F(w2), F(b2), hbuf, (int)MT, DIMD, BIGN);
    ln_elu_kernel<<<(unsigned)MT, 256, 0, stream>>>(hbuf, F(g2), F(be2), nullptr, hbuf, DIMD, 2);
  }

  // ---- 8 sliding-attention layers
  for (int i = 0; i < 8; ++i) {
    const float* attn_w = F(aw)  + (size_t)i * DIMD * HH;
    const float* attn_b = F(ab)  + (size_t)i * HH;
    const float* p1_w   = F(p1w) + (size_t)i * DIMD * HH;
    const float* p1_b   = F(p1b) + (size_t)i * HH;
    const float* pln_g  = F(plg) + (size_t)i * HH;
    const float* pln_b  = F(plb) + (size_t)i * HH;
    const float* p2_w   = F(p2w) + (size_t)i * HH * (WW + 1) * HH;
    const float* p2_b   = F(p2b) + (size_t)i * (WW + 1) * HH;
    const float* v_w    = F(vw)  + (size_t)i * DIMD * DIMD;
    const float* v_b    = F(vb)  + (size_t)i * DIMD;
    const float* vln_g  = F(vlg) + (size_t)i * DIMD;
    const float* vln_b  = F(vlb) + (size_t)i * DIMD;
    const float* o_w    = F(ow)  + (size_t)i * DIMD * DIMD;
    const float* o_b    = F(ob)  + (size_t)i * DIMD;
    const float* oln_g  = F(olg) + (size_t)i * DIMD;
    const float* oln_b  = F(olb) + (size_t)i * DIMD;
    int dil = (i == 0) ? 1 : 3;

    dim3 gA((HH + 63) / 64, MB);
    gemm_bf16_wmma<<<gA, 256, 0, stream>>>(hbuf, attn_w, attn_b, aebuf, (int)MT, HH, DIMD);
    ae_exp_kernel<<<(unsigned)((MT * HH + 255) / 256), 256, 0, stream>>>(aebuf, nmk, (int)(MT * HH));

    dim3 gV((DIMD + 63) / 64, MB);
    gemm_bf16_wmma<<<gV, 256, 0, stream>>>(hbuf, v_w, v_b, big, (int)MT, DIMD, DIMD);
    ln_elu_kernel<<<(unsigned)MT, 256, 0, stream>>>(big, vln_g, vln_b, nullptr, vbuf, DIMD, 0);

    colsum_kernel<<<BB, 128, 0, stream>>>(aebuf, colsb);
    gp_kernel<<<BB, 128, 0, stream>>>(aebuf, vbuf, colsb, gpb);

    dim3 gP1((HH + 63) / 64, 1);
    gemm_bf16_wmma<<<gP1, 256, 0, stream>>>(gpb, p1_w, p1_b, sm1, BB, HH, DIMD);
    ln_elu_kernel<<<BB, 256, 0, stream>>>(sm1, pln_g, pln_b, nullptr, sm2, HH, 0);
    dim3 gP2(((WW + 1) * HH + 63) / 64, 1);
    gemm_bf16_wmma<<<gP2, 256, 0, stream>>>(sm2, p2_w, p2_b, posb, BB, (WW + 1) * HH, HH);
    expv_kernel<<<(BB * (WW + 1) * HH + 255) / 256, 256, 0, stream>>>(posb, BB * (WW + 1) * HH);

    combine_kernel<<<(unsigned)MT, 128, 0, stream>>>(aebuf, vbuf, gpb, posb, obuf, dil);

    gemm_bf16_wmma<<<gV, 256, 0, stream>>>(obuf, o_w, o_b, big, (int)MT, DIMD, DIMD);
    ln_elu_kernel<<<(unsigned)MT, 256, 0, stream>>>(big, oln_g, oln_b, hbuf, hbuf, DIMD, 1);
  }

  // ---- output head -> d_out [12288,60]
  dim3 gO((OUTD + 63) / 64, MB);
  gemm_bf16_wmma<<<gO, 256, 0, stream>>>(hbuf, F(outw), F(outb), (float*)d_out,
                                         (int)MT, OUTD, DIMD);
}
